// Neighbour3dAttnProcessor_25563645346228
// MI455X (gfx1250) — matrix operand order repaired and checked by reading, not compiled, not run on hardware
//
#include <hip/hip_runtime.h>

// ---------------------------------------------------------------------------
// NATTEN 3D neighborhood attention for MI455X (gfx1250), wave32 + WMMA f16.
//   T=16, H=W=32, window 7x7x7, heads=8, head_dim=64, C=512.
// Pipeline: cvt(x)->f16, cvt+T(weights), WMMA GEMM Q/K/V (V stored t-major),
//           flash-style neighborhood attention (all WMMA, shuffle softmax),
//           WMMA GEMM output projection + bias -> f32 d_out.
// ---------------------------------------------------------------------------

typedef __attribute__((ext_vector_type(16))) _Float16 v16h;
typedef __attribute__((ext_vector_type(8)))  float    v8f;

#define T_     16
#define H_     32
#define W_     32
#define HEADS_ 8
#define HD_    64
#define C_     512
#define NTOK   (H_ * W_ * T_)      /* 16384 tokens */
#define NHW    (H_ * W_)           /* 1024 spatial positions */
#define NEG_INF (-3.0e38f)

union TileU {
  v16h h;
  unsigned int u[8];
};

__device__ inline v8f vzero8() {
  v8f z = {0.f, 0.f, 0.f, 0.f, 0.f, 0.f, 0.f, 0.f};
  return z;
}

__device__ inline v8f wmma_f16(v16h a, v16h b, v8f c) {
  // emits v_wmma_f32_16x16x32_f16
  return __builtin_amdgcn_wmma_f32_16x16x32_f16(
      /*neg_a=*/false, a, /*neg_b=*/false, b,
      /*c_mod=*/(short)0, c, /*reuse_a=*/false, /*reuse_b=*/false);
}

// A operand (16x32 f16), ISA layout: lane%16 = row M; VGPR v<4: K=2v+(hi?8:0);
// v>=4: K=16+2(v-4)+(hi?8:0). 'row' is the full per-lane row index.
__device__ inline v16h load_a16x32(const _Float16* __restrict__ base, int row,
                                   int rowStride, int k0, bool hi) {
  TileU t;
  const _Float16* p = base + (size_t)row * rowStride + k0 + (hi ? 8 : 0);
#pragma unroll
  for (int v = 0; v < 4; ++v) t.u[v] = *(const unsigned int*)(p + 2 * v);
#pragma unroll
  for (int v = 4; v < 8; ++v) t.u[v] = *(const unsigned int*)(p + 16 + 2 * (v - 4));
  return t.h;
}

// B operand (32x16 f16): lane%16 = col N; same K order as A: VGPR v<4:
// K=2v+(hi?8:0); v>=4: K=16+2(v-4)+(hi?8:0).
__device__ inline v16h load_b32x16(const _Float16* __restrict__ base, int col,
                                   int colStride, int k0, bool hi) {
  TileU t;
  const _Float16* p = base + (size_t)col * colStride + k0 + (hi ? 8 : 0);
#pragma unroll
  for (int v = 0; v < 4; ++v) t.u[v] = *(const unsigned int*)(p + 2 * v);
#pragma unroll
  for (int v = 4; v < 8; ++v) t.u[v] = *(const unsigned int*)(p + 16 + 2 * (v - 4));
  return t.h;
}

// ------------------------------- prep kernels ------------------------------

__global__ void cvt_f32_f16(const float* __restrict__ in,
                            _Float16* __restrict__ out, int n) {
  int i = blockIdx.x * blockDim.x + threadIdx.x;
  if (i < n) out[i] = (_Float16)in[i];
}

// w [k][n] f32  ->  wt [n][k] f16   (512x512)
__global__ void cvt_transpose_512(const float* __restrict__ w,
                                  _Float16* __restrict__ wt) {
  int i = blockIdx.x * blockDim.x + threadIdx.x;  // k*512 + n
  int k = i >> 9, n = i & 511;
  wt[(size_t)n * 512 + k] = (_Float16)w[i];
}

// ------------------------------- GEMM kernel -------------------------------
// Out = Ah[M x 512] * W, W supplied transposed as wt[n][k] f16.
// grid.x = M/16 (m-tile), 8 waves/block each owning 64 columns.
// mode 0: f16 out[m][n]         (Q, K)
// mode 1: f16 V transposed: vt[head][hw][d][t]   (for AV B-operand)
// mode 2: f32 out[m][n] + bias  (final projection)
__global__ __launch_bounds__(256) void gemm_wmma(
    const _Float16* __restrict__ ah, const _Float16* __restrict__ wt,
    _Float16* __restrict__ outh, float* __restrict__ outf,
    const float* __restrict__ bias, int mode) {
  const int lane = threadIdx.x & 31;
  const int wave = threadIdx.x >> 5;  // 0..7 -> n-tile of 64
  const int lr = lane & 15;
  const bool hi = lane >= 16;
  const int mt = blockIdx.x;
  const int m0 = mt * 16;
  const int n0 = wave * 64;

  v8f acc[4] = {vzero8(), vzero8(), vzero8(), vzero8()};

  for (int k0 = 0; k0 < C_; k0 += 32) {
    v16h a = load_a16x32(ah, m0 + lr, C_, k0, hi);
#pragma unroll
    for (int nt = 0; nt < 4; ++nt) {
      v16h b = load_b32x16(wt, n0 + nt * 16 + lr, C_, k0, hi);
      acc[nt] = wmma_f16(a, b, acc[nt]);
    }
  }

  // D layout: VGPR g -> m = m0 + g + 8*hi ; lane -> n = n0 + nt*16 + lr
#pragma unroll
  for (int nt = 0; nt < 4; ++nt) {
#pragma unroll
    for (int g = 0; g < 8; ++g) {
      const int m = m0 + g + (hi ? 8 : 0);
      const int n = n0 + nt * 16 + lr;
      const float val = acc[nt][g];
      if (mode == 0) {
        outh[(size_t)m * C_ + n] = (_Float16)val;
      } else if (mode == 1) {
        const int head = n >> 6, dd = n & 63, tt = g + (hi ? 8 : 0);
        outh[(((size_t)head * NHW + mt) * HD_ + dd) * T_ + tt] = (_Float16)val;
      } else {
        outf[(size_t)m * C_ + n] = val + bias[n];
      }
    }
  }
}

// ----------------------------- attention kernel ----------------------------
// block = one spatial position hw (1024 blocks), wave = one head (8 waves).
// Tile = 16 temporal queries; keys = 49 clamped spatial neighbors x 16 t',
// temporal-window mask reproduces the exact 343-neighbor NATTEN set.
// S^T = K * Q^T via WMMA; S^T D-layout packs directly into the AV A-operand.
__global__ __launch_bounds__(256) void natten_attn(
    const _Float16* __restrict__ qh, const _Float16* __restrict__ kh,
    const _Float16* __restrict__ vt, _Float16* __restrict__ oh) {
  const int hw = blockIdx.x;
  const int h = hw >> 5, w = hw & 31;
  const int hh = threadIdx.x >> 5;  // head
  const int lane = threadIdx.x & 31;
  const int lr = lane & 15;
  const bool hi = lane >= 16;

  // temporal window for query t = lr (clamped, dilation 1)
  const int ts = min(max(lr - 3, 0), T_ - 7);
  const int te = ts + 6;

  // Q as B operand (lane = query, K = head dims), reused for all 49 tiles
  const _Float16* qbase = qh + (size_t)(hw * T_) * C_ + hh * HD_;
  const v16h qb0 = load_b32x16(qbase, lr, C_, 0, hi);
  const v16h qb1 = load_b32x16(qbase, lr, C_, 32, hi);

  v8f acc[4] = {vzero8(), vzero8(), vzero8(), vzero8()};
  float m_run = NEG_INF, l_run = 0.f;

  const int hs = min(max(h - 3, 0), H_ - 7);
  const int ws = min(max(w - 3, 0), W_ - 7);

  for (int ip = 0; ip < 25; ++ip) {  // 49 spatial tiles as 24 pairs + 1 dup
    const int s0i = 2 * ip;
    int s1i = 2 * ip + 1;
    const bool valid1 = (s1i < 49);
    if (!valid1) s1i = s0i;
    const int n0hw = (hs + s0i / 7) * W_ + (ws + s0i % 7);
    const int n1hw = (hs + s1i / 7) * W_ + (ws + s1i % 7);

    // S^T tiles: [16 keys x 16 queries], A = K tile, B = Q^T
    v8f s0 = vzero8(), s1 = vzero8();
    {
      const _Float16* kb0 = kh + (size_t)(n0hw * T_) * C_ + hh * HD_;
      s0 = wmma_f16(load_a16x32(kb0, lr, C_, 0, hi), qb0, s0);
      s0 = wmma_f16(load_a16x32(kb0, lr, C_, 32, hi), qb1, s0);
      const _Float16* kb1 = kh + (size_t)(n1hw * T_) * C_ + hh * HD_;
      s1 = wmma_f16(load_a16x32(kb1, lr, C_, 0, hi), qb0, s1);
      s1 = wmma_f16(load_a16x32(kb1, lr, C_, 32, hi), qb1, s1);
    }

    // temporal mask: key t' = g + 8*hi, query t = lr
#pragma unroll
    for (int g = 0; g < 8; ++g) {
      const int tp = g + (hi ? 8 : 0);
      const bool ok = (tp >= ts) && (tp <= te);
      s0[g] = ok ? s0[g] : NEG_INF;
      s1[g] = (ok && valid1) ? s1[g] : NEG_INF;
    }

    // online softmax (per-query rows live on lanes; halves hold key 0-7/8-15)
    float tmax = NEG_INF;
#pragma unroll
    for (int g = 0; g < 8; ++g) {
      tmax = fmaxf(tmax, s0[g]);
      tmax = fmaxf(tmax, s1[g]);
    }
    tmax = fmaxf(tmax, __shfl_xor(tmax, 16, 32));
    const float m_new = fmaxf(m_run, tmax);
    const float alpha = __expf(m_run - m_new);
    float psum = 0.f;
#pragma unroll
    for (int g = 0; g < 8; ++g) {
      s0[g] = __expf(s0[g] - m_new);
      s1[g] = __expf(s1[g] - m_new);
      psum += s0[g] + s1[g];
    }
    psum += __shfl_xor(psum, 16, 32);
    l_run = l_run * alpha + psum;
    m_run = m_new;

    // pack P (f32 S^T D-layout) straight into the f16 A-operand layout:
    // A VGPR v<4 = pair(s0[2v], s0[2v+1]); v>=4 = pair(s1[2(v-4)], ...)
    TileU pa;
#pragma unroll
    for (int j = 0; j < 4; ++j) {
      pa.h[2 * j]     = (_Float16)s0[2 * j];
      pa.h[2 * j + 1] = (_Float16)s0[2 * j + 1];
      pa.h[8 + 2 * j] = (_Float16)s1[2 * j];
      pa.h[9 + 2 * j] = (_Float16)s1[2 * j + 1];
    }

    // rescale accumulators by alpha[query], query = g + 8*hi (ds_bpermute)
#pragma unroll
    for (int g = 0; g < 8; ++g) {
      const float ag = __shfl(alpha, g + (hi ? 8 : 0), 32);
#pragma unroll
      for (int nt = 0; nt < 4; ++nt) acc[nt][g] *= ag;
    }

    // AV: B = V tile [32 keys x 16 dims]; VGPR v<4 -> nbr0 t'=2v+8*hi,
    // v>=4 -> nbr1 t'=2(v-4)+8*hi; vt is t-contiguous so pairs are dwords.
    const _Float16* vb0 = vt + (((size_t)hh * NHW + n0hw) * HD_) * T_ + (hi ? 8 : 0);
    const _Float16* vb1 = vt + (((size_t)hh * NHW + n1hw) * HD_) * T_ + (hi ? 8 : 0);
#pragma unroll
    for (int nt = 0; nt < 4; ++nt) {
      const int d = nt * 16 + lr;
      TileU b;
      const _Float16* p0 = vb0 + (size_t)d * T_;
      const _Float16* p1 = vb1 + (size_t)d * T_;
#pragma unroll
      for (int v = 0; v < 4; ++v) b.u[v] = *(const unsigned int*)(p0 + 2 * v);
#pragma unroll
      for (int v = 4; v < 8; ++v) b.u[v] = *(const unsigned int*)(p1 + 2 * (v - 4));
      acc[nt] = wmma_f16(pa.h, b.h, acc[nt]);
    }
  }

  // normalize by row sums and store f16 O in [token][C] layout
#pragma unroll
  for (int g = 0; g < 8; ++g) {
    const float lq = __shfl(l_run, g + (hi ? 8 : 0), 32);
    const float inv = 1.0f / lq;
    const int tt = g + (hi ? 8 : 0);
#pragma unroll
    for (int nt = 0; nt < 4; ++nt) {
      oh[(size_t)(hw * T_ + tt) * C_ + hh * HD_ + nt * 16 + lr] =
          (_Float16)(acc[nt][g] * inv);
    }
  }
}

// --------------------------------- launch ----------------------------------

extern "C" void kernel_launch(void* const* d_in, const int* in_sizes, int n_in,
                              void* d_out, int out_size, void* d_ws,
                              size_t ws_size, hipStream_t stream) {
  const float* x   = (const float*)d_in[0];  // [16384, 512]
  const float* w_q = (const float*)d_in[1];  // [512, 512]
  const float* w_k = (const float*)d_in[2];
  const float* w_v = (const float*)d_in[3];
  const float* w_o = (const float*)d_in[4];
  const float* b_o = (const float*)d_in[5];  // [512]
  float* out = (float*)d_out;

  // workspace carving (~86 MB total)
  char* ws = (char*)d_ws;
  const size_t TOK_BYTES = (size_t)NTOK * C_ * sizeof(_Float16);  // 16 MB
  const size_t W_BYTES = (size_t)C_ * C_ * sizeof(_Float16);      // 512 KB
  _Float16* xh  = (_Float16*)ws;              ws += TOK_BYTES;
  _Float16* qhb = (_Float16*)ws;              ws += TOK_BYTES;
  _Float16* khb = (_Float16*)ws;              ws += TOK_BYTES;
  _Float16* vtb = (_Float16*)ws;              ws += TOK_BYTES;
  _Float16* ohb = (_Float16*)ws;              ws += TOK_BYTES;
  _Float16* wqT = (_Float16*)ws;              ws += W_BYTES;
  _Float16* wkT = (_Float16*)ws;              ws += W_BYTES;
  _Float16* wvT = (_Float16*)ws;              ws += W_BYTES;
  _Float16* woT = (_Float16*)ws;              ws += W_BYTES;

  // prep: f16 conversion + weight transposes
  cvt_f32_f16<<<(NTOK * C_ + 255) / 256, 256, 0, stream>>>(x, xh, NTOK * C_);
  cvt_transpose_512<<<(C_ * C_) / 256, 256, 0, stream>>>(w_q, wqT);
  cvt_transpose_512<<<(C_ * C_) / 256, 256, 0, stream>>>(w_k, wkT);
  cvt_transpose_512<<<(C_ * C_) / 256, 256, 0, stream>>>(w_v, wvT);
  cvt_transpose_512<<<(C_ * C_) / 256, 256, 0, stream>>>(w_o, woT);

  // QKV projections (V stored t-major per head for the AV B operand)
  gemm_wmma<<<NTOK / 16, 256, 0, stream>>>(xh, wqT, qhb, nullptr, nullptr, 0);
  gemm_wmma<<<NTOK / 16, 256, 0, stream>>>(xh, wkT, khb, nullptr, nullptr, 0);
  gemm_wmma<<<NTOK / 16, 256, 0, stream>>>(xh, wvT, vtb, nullptr, nullptr, 1);

  // neighborhood attention: 1024 spatial blocks x 8 head-waves
  natten_attn<<<NHW, 256, 0, stream>>>(qhb, khb, vtb, ohb);

  // output projection + bias -> f32
  gemm_wmma<<<NTOK / 16, 256, 0, stream>>>(ohb, woT, nullptr, out, b_o, 2);
}
